// GNN_5866925326816
// MI455X (gfx1250) — matrix operand order repaired and checked by reading, not compiled, not run on hardware
//
#include <hip/hip_runtime.h>

typedef __attribute__((ext_vector_type(16))) __bf16 v16bf;
typedef __attribute__((ext_vector_type(8)))  float  v8f;

using u16 = unsigned short;
using u32 = unsigned int;

constexpr int NN   = 50000;
constexpr int EE   = 600000;
constexpr int FIN  = 128;
constexpr int FE   = 32;
constexpr int HH   = 128;
constexpr int KTOP = NN / 2;   // 25000

__device__ __forceinline__ u16 f2bf(float x) {
    u32 u = __builtin_bit_cast(u32, x);
    return (u16)((u + 0x7FFFu + ((u >> 16) & 1u)) >> 16);
}

union ABFrag { v16bf v; u16 u[16]; uint4 q[2]; };

// ---- CDNA5 async global->LDS DMA helpers (ASYNCcnt-tracked) ----
__device__ __forceinline__ u32 lds_off(const void* p) {
    return (u32)(unsigned long long)p;   // low 32 bits of generic shared ptr = LDS offset
}
__device__ __forceinline__ void async_b128(u32 lds, const void* g) {
    asm volatile("global_load_async_to_lds_b128 %0, %1, off"
                 :: "v"(lds), "v"(g) : "memory");
}
__device__ __forceinline__ void wait_async() {
    asm volatile("s_wait_asynccnt 0x0" ::: "memory");
}

// ---------------------------------------------------------------- utilities

__global__ void cast_bf16_kernel(const float* __restrict__ s, u16* __restrict__ d, long n) {
    long i  = (long)blockIdx.x * blockDim.x + threadIdx.x;
    long st = (long)gridDim.x * blockDim.x;
    for (; i < n; i += st) d[i] = f2bf(s[i]);
}

// Repack row-major f32 weights [KDIM][FOUT] into bf16 B-fragment order:
// dst[((kc*CT + t)*32 + lane)*16 + i] = W[kc*32 + (lane>>4)*16 + i][t*16 + (lane&15)]
__global__ void pack_w_kernel(const float* __restrict__ W, u16* __restrict__ dst,
                              int KDIM, int FOUT) {
    int  CT    = FOUT / 16;
    long total = (long)KDIM * FOUT;
    long idx   = (long)blockIdx.x * blockDim.x + threadIdx.x;
    long st    = (long)gridDim.x * blockDim.x;
    for (; idx < total; idx += st) {
        int  i    = (int)(idx & 15);
        int  lane = (int)((idx >> 4) & 31);
        long tile = idx >> 9;
        int  t    = (int)(tile % CT);
        int  kc   = (int)(tile / CT);
        int  n    = lane & 15;
        int  half = lane >> 4;
        int  k    = kc * 32 + half * 16 + i;
        int  c    = t * 16 + n;
        dst[idx] = f2bf(W[(long)k * FOUT + c]);
    }
}

__global__ void zero_kernel(float* __restrict__ p, long n) {
    long i  = (long)blockIdx.x * blockDim.x + threadIdx.x;
    long st = (long)gridDim.x * blockDim.x;
    for (; i < n; i += st) p[i] = 0.f;
}

__global__ void invnorm_kernel(const float* __restrict__ w, float* __restrict__ out) {
    __shared__ float red[128];
    float v = w[threadIdx.x];
    red[threadIdx.x] = v * v;
    __syncthreads();
    for (int s = 64; s > 0; s >>= 1) {
        if ((int)threadIdx.x < s) red[threadIdx.x] += red[threadIdx.x + s];
        __syncthreads();
    }
    if (threadIdx.x == 0) out[0] = rsqrtf(red[0]);
}

// score = tanh(dot(x1_i, pool_w) * invnorm); key = order-preserving uint of score
__global__ void score_kernel(const float* __restrict__ x1, const float* __restrict__ pw,
                             const float* __restrict__ invn, float* __restrict__ score,
                             u32* __restrict__ keys, int n) {
    int node = blockIdx.x * 8 + (threadIdx.x >> 5);
    int lane = threadIdx.x & 31;
    if (node >= n) return;
    const float* row = x1 + (long)node * HH;
    float s = 0.f;
    for (int c = lane; c < HH; c += 32) s += row[c] * pw[c];
    for (int off = 16; off > 0; off >>= 1) s += __shfl_down(s, off, 32);
    if (lane == 0) {
        float sc = tanhf(s * invn[0]);
        score[node] = sc;
        u32 u = __builtin_bit_cast(u32, sc);
        u = (u & 0x80000000u) ? ~u : (u | 0x80000000u);
        keys[node] = u;
    }
}

// rank_i = #{j : key_j > key_i} + #{j < i : key_j == key_i}  (stable descending rank)
__global__ void rank_kernel(const u32* __restrict__ keys, int* __restrict__ rank, int n) {
    __shared__ u32 sk[1024];
    int gid = blockIdx.x * 256 + threadIdx.x;
    u32 my = (gid < n) ? keys[gid] : 0u;
    int cnt = 0;
    for (int base = 0; base < n; base += 1024) {
        int lim = min(1024, n - base);
        __syncthreads();
        for (int i = threadIdx.x; i < lim; i += 256) sk[i] = keys[base + i];
        __syncthreads();
        if (gid < n) {
            for (int i = 0; i < lim; ++i) {
                u32 k = sk[i];
                cnt += (int)((k > my) || (k == my && (base + i) < gid));
            }
        }
    }
    if (gid < n) rank[gid] = cnt;
}

// xp = sel ? x1 * score : 0   (bf16)
__global__ void gate_kernel(const float* __restrict__ x1, const float* __restrict__ score,
                            const int* __restrict__ rank, u16* __restrict__ xp, long n) {
    long i  = (long)blockIdx.x * blockDim.x + threadIdx.x;
    long st = (long)gridDim.x * blockDim.x;
    for (; i < n; i += st) {
        int node = (int)(i >> 7);   // HH = 128
        u16 v = 0;
        if (rank[node] < KTOP) v = f2bf(x1[i] * score[node]);
        xp[i] = v;
    }
}

// mode 0: identity rows, all valid. mode 1: original rows, masked by sel.
// mode 2: rank-relabeled rows, masked by sel (faithful to reference's reuse of pooled indices).
__global__ void build_edges_kernel(const long long* __restrict__ ei, const int* __restrict__ rank,
                                   int mode, int* __restrict__ esrc, int* __restrict__ edst, int ne) {
    int i  = blockIdx.x * blockDim.x + threadIdx.x;
    int st = gridDim.x * blockDim.x;
    for (; i < ne; i += st) {
        int s = (int)ei[i];
        int d = (int)ei[(long)ne + i];
        if (mode == 0) { esrc[i] = s; edst[i] = d; continue; }
        bool ok = (rank[s] < KTOP) && (rank[d] < KTOP);
        if (!ok)            { esrc[i] = -1;       edst[i] = -1; }
        else if (mode == 1) { esrc[i] = s;        edst[i] = d; }
        else                { esrc[i] = rank[s];  edst[i] = rank[d]; }
    }
}

// ---------------------------------------------------------------- WMMA edge messages
// m_e = relu([feat[srcrow], ea_e] @ W + b); atomicAdd into agg[dstrow] (skipped if masked).
// Whole packed weight resident in LDS; A chunks DMA'd via async global->LDS, double-buffered.
template<int KDIM, int FOUT>
__global__ void __launch_bounds__(256)
edge_msg_kernel(const u16* __restrict__ feat, const u16* __restrict__ ea,
                const int* __restrict__ esrc, const int* __restrict__ edst,
                const u16* __restrict__ Wpk, const float* __restrict__ bias,
                float* __restrict__ agg, int ne) {
    constexpr int F1 = KDIM - FE;
    constexpr int KC = KDIM / 32;
    constexpr int CT = FOUT / 16;
    constexpr int EB = 128;
    __shared__ __align__(16) u16 sW[KDIM * FOUT];   // full pre-swizzled weights (<=104KB)
    __shared__ __align__(16) u16 sA[2][EB * 32];    // double-buffered A chunks
    __shared__ int sRow[EB];
    __shared__ int sDst[EB];

    const int wave = threadIdx.x >> 5;
    const int lane = threadIdx.x & 31;
    const int lm   = lane & 15;
    const int half = lane >> 4;
    const int eloc = threadIdx.x >> 1;   // staging: edge slot (0..127)
    const int hloc = threadIdx.x & 1;    // staging: 16-element half

    // one-time whole-weight load: async DMA straight into LDS
    for (int i = threadIdx.x; i < KDIM * FOUT / 8; i += 256)
        async_b128(lds_off(&sW[i * 8]), Wpk + i * 8);
    wait_async();
    __syncthreads();

    for (long base = (long)blockIdx.x * EB; base < ne; base += (long)gridDim.x * EB) {
        for (int i = threadIdx.x; i < EB; i += 256) {
            long e = base + i;
            sRow[i] = (e < ne) ? esrc[e] : -1;
            sDst[i] = (e < ne) ? edst[e] : -1;
        }
        __syncthreads();

        // stage chunk 0 (per-lane async DMA; masked/tail lanes skip -> rows never stored)
        {
            long ge = base + eloc;
            int  c0 = hloc * 16;
            u32  ld = lds_off(&sA[0][eloc * 32 + hloc * 16]);
            if (ge < ne) {
                if (c0 < F1) {   // always true for chunk 0 (F1 >= 32)
                    int row = sRow[eloc];
                    if (row >= 0) async_b128(ld, feat + (long)row * F1 + c0);
                } else {
                    async_b128(ld, ea + ge * FE + (c0 - F1));
                }
            }
        }
        wait_async();
        __syncthreads();

        v8f acc[CT];
        #pragma unroll
        for (int t = 0; t < CT; ++t)
            #pragma unroll
            for (int r = 0; r < 8; ++r) acc[t][r] = 0.f;

        for (int kc = 0; kc < KC; ++kc) {
            // prefetch chunk kc+1 into the other buffer (overlaps with WMMA below)
            if (kc + 1 < KC) {
                long ge = base + eloc;
                int  c0 = (kc + 1) * 32 + hloc * 16;
                u32  ld = lds_off(&sA[(kc + 1) & 1][eloc * 32 + hloc * 16]);
                if (ge < ne) {
                    if (c0 < F1) {
                        int row = sRow[eloc];
                        if (row >= 0) async_b128(ld, feat + (long)row * F1 + c0);
                    } else {
                        async_b128(ld, ea + ge * FE + (c0 - F1));
                    }
                }
            }

            // A fragment: one contiguous 32B run per lane (contiguous k order, matches packed B)
            ABFrag af;
            const u16* arow = &sA[kc & 1][(wave * 16 + lm) * 32];
            af.q[0] = ((const uint4*)arow)[2 * half];      // arow[half*16 .. half*16+7]
            af.q[1] = ((const uint4*)arow)[2 * half + 1];  // arow[half*16+8 .. +7]

            #pragma unroll
            for (int t = 0; t < CT; ++t) {
                ABFrag bfr;
                const uint4* bp = (const uint4*)(sW + ((kc * CT + t) * 32 + lane) * 16);
                bfr.q[0] = bp[0];
                bfr.q[1] = bp[1];
                acc[t] = __builtin_amdgcn_wmma_f32_16x16x32_bf16(
                    false, af.v, false, bfr.v, (short)0, acc[t], false, false);
            }
            wait_async();      // chunk kc+1 DMA complete (this wave's)
            __syncthreads();   // all waves done reading buf[kc&1] + all DMAs visible
        }

        // store: relu(acc + bias) scatter-added per destination row
        #pragma unroll
        for (int t = 0; t < CT; ++t) {
            int   nc = t * 16 + lm;
            float b  = bias[nc];
            #pragma unroll
            for (int r = 0; r < 8; ++r) {
                int  el = wave * 16 + r + half * 8;
                long ge = base + el;
                if (ge < ne) {
                    int d = sDst[el];
                    if (d >= 0) {
                        float v = acc[t][r] + b;
                        v = v > 0.f ? v : 0.f;
                        atomicAdd(agg + (long)d * FOUT + nc, v);
                    }
                }
            }
        }
        __syncthreads();
    }
}

// ---------------------------------------------------------------- WMMA node update
// out = relu([in1 (bf16,F1) | agg (f32,F2)] @ W + b), optional sel-mask, f32 and/or bf16 outputs.
template<int KDIM, int FOUT, int F1>
__global__ void __launch_bounds__(256)
node_update_kernel(const u16* __restrict__ in1, const float* __restrict__ agg,
                   const u16* __restrict__ Wpk, const float* __restrict__ bias,
                   float* __restrict__ out_f32, u16* __restrict__ out_bf, int ldo,
                   const int* __restrict__ rank, int nrows) {
    constexpr int F2 = KDIM - F1;
    constexpr int KC = KDIM / 32;
    constexpr int CT = FOUT / 16;
    constexpr int RB = 128;
    __shared__ __align__(16) u16 sA[RB * 32];
    __shared__ __align__(16) u16 sB[32 * FOUT];

    const int wave = threadIdx.x >> 5;
    const int lane = threadIdx.x & 31;
    const int lm   = lane & 15;
    const int half = lane >> 4;
    const long base = (long)blockIdx.x * RB;

    v8f acc[CT];
    #pragma unroll
    for (int t = 0; t < CT; ++t)
        #pragma unroll
        for (int r = 0; r < 8; ++r) acc[t][r] = 0.f;

    for (int kc = 0; kc < KC; ++kc) {
        {   // stage A chunk: 128 rows x 32 cols
            int  i   = threadIdx.x >> 1;
            int  h   = threadIdx.x & 1;
            long row = base + i;
            int  c0  = kc * 32 + h * 16;
            u16* s   = sA + i * 32 + h * 16;
            if (row < nrows) {
                if (c0 < F1) {
                    const uint4* p = (const uint4*)(in1 + row * F1 + c0);
                    ((uint4*)s)[0] = p[0];
                    ((uint4*)s)[1] = p[1];
                } else {
                    const float* p = agg + row * F2 + (c0 - F1);
                    #pragma unroll
                    for (int j = 0; j < 16; ++j) s[j] = f2bf(p[j]);
                }
            } else {
                ((uint4*)s)[0] = make_uint4(0, 0, 0, 0);
                ((uint4*)s)[1] = make_uint4(0, 0, 0, 0);
            }
        }
        for (int i = threadIdx.x; i < 32 * FOUT / 8; i += 256)
            ((uint4*)sB)[i] = ((const uint4*)(Wpk + (long)kc * 32 * FOUT))[i];
        __syncthreads();

        ABFrag af;
        const u16* arow = sA + (wave * 16 + lm) * 32;
        af.q[0] = ((const uint4*)arow)[2 * half];
        af.q[1] = ((const uint4*)arow)[2 * half + 1];

        #pragma unroll
        for (int t = 0; t < CT; ++t) {
            ABFrag bfr;
            const uint4* bp = (const uint4*)(sB + (t * 32 + lane) * 16);
            bfr.q[0] = bp[0];
            bfr.q[1] = bp[1];
            acc[t] = __builtin_amdgcn_wmma_f32_16x16x32_bf16(
                false, af.v, false, bfr.v, (short)0, acc[t], false, false);
        }
        __syncthreads();
    }

    #pragma unroll
    for (int t = 0; t < CT; ++t) {
        int   nc = t * 16 + lm;
        float b  = bias[nc];
        #pragma unroll
        for (int r = 0; r < 8; ++r) {
            long row = base + wave * 16 + r + half * 8;
            if (row < nrows) {
                float v = acc[t][r] + b;
                v = v > 0.f ? v : 0.f;
                if (rank && rank[row] >= KTOP) v = 0.f;   // unpool mask
                if (out_f32) out_f32[row * FOUT + nc] = v;
                if (out_bf)  out_bf[row * (long)ldo + nc] = f2bf(v);
            }
        }
    }
}

// ---------------------------------------------------------------- host orchestration

extern "C" void kernel_launch(void* const* d_in, const int* in_sizes, int n_in,
                              void* d_out, int out_size, void* d_ws, size_t ws_size,
                              hipStream_t stream) {
    (void)in_sizes; (void)n_in; (void)out_size; (void)ws_size;

    const float* x    = (const float*)d_in[0];
    const float* ea   = (const float*)d_in[1];
    const float* We_d = (const float*)d_in[2];  const float* be_d = (const float*)d_in[3];
    const float* Wn_d = (const float*)d_in[4];  const float* bn_d = (const float*)d_in[5];
    const float* poolw= (const float*)d_in[6];
    const float* We_b = (const float*)d_in[7];  const float* be_b = (const float*)d_in[8];
    const float* Wn_b = (const float*)d_in[9];  const float* bn_b = (const float*)d_in[10];
    const float* We_u = (const float*)d_in[11]; const float* be_u = (const float*)d_in[12];
    const float* Wn_u = (const float*)d_in[13]; const float* bn_u = (const float*)d_in[14];
    const float* We_f = (const float*)d_in[15]; const float* be_f = (const float*)d_in[16];
    const float* Wn_f = (const float*)d_in[17]; const float* bn_f = (const float*)d_in[18];
    const long long* eidx = (const long long*)d_in[19];
    float* out = (float*)d_out;

    // workspace carve (256B aligned)
    char* wp = (char*)d_ws;
    auto carve = [&](size_t bytes) -> char* {
        char* p = wp;
        wp += (bytes + 255) & ~(size_t)255;
        return p;
    };
    u16* ea_bf   = (u16*)carve((size_t)EE * FE * 2);
    u16* x_bf    = (u16*)carve((size_t)NN * FIN * 2);
    u16* We_d_pk = (u16*)carve(160 * 128 * 2);
    u16* Wn_d_pk = (u16*)carve(256 * 128 * 2);
    u16* We_b_pk = (u16*)carve(160 * 256 * 2);
    u16* Wn_b_pk = (u16*)carve(384 * 256 * 2);
    u16* We_u_pk = (u16*)carve(416 * 128 * 2);
    u16* Wn_u_pk = (u16*)carve(512 * 128 * 2);
    u16* We_f_pk = (u16*)carve(160 * 128 * 2);
    u16* Wn_f_pk = (u16*)carve(256 * 128 * 2);
    float* x1    = (float*)carve((size_t)NN * HH * 4);
    float* agg   = (float*)carve((size_t)NN * 256 * 4);
    float* score = (float*)carve((size_t)NN * 4);
    u32*   keys  = (u32*)carve((size_t)NN * 4);
    int*   rank  = (int*)carve((size_t)NN * 4);
    float* invn  = (float*)carve(256);
    u16* xp_bf   = (u16*)carve((size_t)NN * HH * 2);
    u16* xc_bf   = (u16*)carve((size_t)NN * 384 * 2);
    u16* x3_bf   = (u16*)carve((size_t)NN * HH * 2);
    int* esrc    = (int*)carve((size_t)EE * 4);
    int* edst    = (int*)carve((size_t)EE * 4);

    const int GS = 1024;      // grid-stride helper grid
    const int EG = 2048;      // edge kernel grid
    const int NG = (NN + 127) / 128;

    // precision conversions + weight repack into B-fragment order
    cast_bf16_kernel<<<GS, 256, 0, stream>>>(x,  x_bf,  (long)NN * FIN);
    cast_bf16_kernel<<<GS, 256, 0, stream>>>(ea, ea_bf, (long)EE * FE);
    pack_w_kernel<<<GS, 256, 0, stream>>>(We_d, We_d_pk, 160, 128);
    pack_w_kernel<<<GS, 256, 0, stream>>>(Wn_d, Wn_d_pk, 256, 128);
    pack_w_kernel<<<GS, 256, 0, stream>>>(We_b, We_b_pk, 160, 256);
    pack_w_kernel<<<GS, 256, 0, stream>>>(Wn_b, Wn_b_pk, 384, 256);
    pack_w_kernel<<<GS, 256, 0, stream>>>(We_u, We_u_pk, 416, 128);
    pack_w_kernel<<<GS, 256, 0, stream>>>(Wn_u, Wn_u_pk, 512, 128);
    pack_w_kernel<<<GS, 256, 0, stream>>>(We_f, We_f_pk, 160, 128);
    pack_w_kernel<<<GS, 256, 0, stream>>>(Wn_f, Wn_f_pk, 256, 128);
    invnorm_kernel<<<1, 128, 0, stream>>>(poolw, invn);

    // ---- down conv: x1 = conv(x)
    build_edges_kernel<<<GS, 256, 0, stream>>>(eidx, nullptr, 0, esrc, edst, EE);
    zero_kernel<<<GS, 256, 0, stream>>>(agg, (long)NN * 128);
    edge_msg_kernel<160, 128><<<EG, 256, 0, stream>>>(x_bf, ea_bf, esrc, edst,
                                                      We_d_pk, be_d, agg, EE);
    // f32 -> x1 ; bf16 -> xc columns 256..383 (skip connection half of xc)
    node_update_kernel<256, 128, 128><<<NG, 256, 0, stream>>>(
        x_bf, agg, Wn_d_pk, bn_d, x1, xc_bf + 256, 384, nullptr, NN);

    // ---- TopK pooling: scores, stable descending ranks, gated features
    score_kernel<<<(NN + 7) / 8, 256, 0, stream>>>(x1, poolw, invn, score, keys, NN);
    rank_kernel<<<(NN + 255) / 256, 256, 0, stream>>>(keys, rank, NN);
    gate_kernel<<<GS, 256, 0, stream>>>(x1, score, rank, xp_bf, (long)NN * HH);

    // ---- bottleneck conv (original index space, sel-masked); bf16 out -> xc cols 0..255
    build_edges_kernel<<<GS, 256, 0, stream>>>(eidx, rank, 1, esrc, edst, EE);
    zero_kernel<<<GS, 256, 0, stream>>>(agg, (long)NN * 256);
    edge_msg_kernel<160, 256><<<EG, 256, 0, stream>>>(xp_bf, ea_bf, esrc, edst,
                                                      We_b_pk, be_b, agg, EE);
    node_update_kernel<384, 256, 128><<<NG, 256, 0, stream>>>(
        xp_bf, agg, Wn_b_pk, bn_b, nullptr, xc_bf, 384, rank, NN);

    // ---- up conv: faithful relabeled-index reuse (rows = rank[src]/rank[dst])
    build_edges_kernel<<<GS, 256, 0, stream>>>(eidx, rank, 2, esrc, edst, EE);
    zero_kernel<<<GS, 256, 0, stream>>>(agg, (long)NN * 128);
    edge_msg_kernel<416, 128><<<EG, 256, 0, stream>>>(xc_bf, ea_bf, esrc, edst,
                                                      We_u_pk, be_u, agg, EE);
    node_update_kernel<512, 128, 384><<<NG, 256, 0, stream>>>(
        xc_bf, agg, Wn_u_pk, bn_u, nullptr, x3_bf, 128, nullptr, NN);

    // ---- final conv (same relabeled edge table)
    zero_kernel<<<GS, 256, 0, stream>>>(agg, (long)NN * 128);
    edge_msg_kernel<160, 128><<<EG, 256, 0, stream>>>(x3_bf, ea_bf, esrc, edst,
                                                      We_f_pk, be_f, agg, EE);
    node_update_kernel<256, 128, 128><<<NG, 256, 0, stream>>>(
        x3_bf, agg, Wn_f_pk, bn_f, out, nullptr, 0, nullptr, NN);
}
